// amortized_fit_66683662238249
// MI455X (gfx1250) — matrix operand order repaired and checked by reading, not compiled, not run on hardware
//
#include <hip/hip_runtime.h>
#include <hip/hip_bf16.h>
#include <math.h>

#define HDIM 512
#define NWG  32
#define NTHR 256
#define NKT1 34          // phase-1 K tiles (K padded to 1088)
#define NRT1 128         // 2048 rows / 16
#define NKT2 16          // phase-2 K tiles (K = 512)
#define NRT2 96          // 1536 rows / 16
#define KP1  (NKT1 * 32) // 1088

typedef __attribute__((ext_vector_type(16))) __bf16 v16bf;
typedef __attribute__((ext_vector_type(8)))  float  v8f;

// ---------------- workspace layout (bytes) ----------------
constexpr size_t OFF_V1G   = 64;                     // float[1088]  [x(514); h(512); pad]
constexpr size_t OFF_CBUF  = OFF_V1G  + 1088 * 4;    // float[1024]  double-buffered c
constexpr size_t OFF_GATES = OFF_CBUF + 1024 * 4;    // float[2048]
constexpr size_t OFF_BSUM  = OFF_GATES + 2048 * 4;   // float[2048]  b_ih + b_hh
constexpr size_t OFF_BIASN = OFF_BSUM + 2048 * 4;    // float[1536]  [b_nh1; b_nh2; b_nin1]
constexpr size_t OFF_ZRAW  = OFF_BIASN + 1536 * 4;   // float[32768]
constexpr size_t OFF_P1    = ((OFF_ZRAW + 32768 * 4 + 63) / 64) * 64; // bf16 packed phase-1 A
constexpr size_t P1_ELEMS  = (size_t)NRT1 * NKT1 * 512;
constexpr size_t OFF_P2    = OFF_P1 + P1_ELEMS * 2;                   // bf16 packed phase-2 A
constexpr size_t P2_ELEMS  = (size_t)NRT2 * NKT2 * 512;

// ---------------- helpers ----------------
__device__ __forceinline__ float sigm(float x) { return 1.0f / (1.0f + expf(-x)); }

__device__ __forceinline__ unsigned fenc(float f) {  // monotone float->uint for atomic max
  unsigned u = __float_as_uint(f);
  return (u & 0x80000000u) ? ~u : (u | 0x80000000u);
}
__device__ __forceinline__ float fdec(unsigned e) {
  return __uint_as_float((e & 0x80000000u) ? (e & 0x7fffffffu) : ~e);
}

__device__ __forceinline__ void gbar(unsigned* bar, unsigned target) {
  __threadfence();
  __syncthreads();
  if (threadIdx.x == 0) {
    __hip_atomic_fetch_add(bar, 1u, __ATOMIC_RELEASE, __HIP_MEMORY_SCOPE_AGENT);
    while (__hip_atomic_load(bar, __ATOMIC_ACQUIRE, __HIP_MEMORY_SCOPE_AGENT) < target) {
      __builtin_amdgcn_s_sleep(1);   // ~64 clk: fine-grained to not inflate barrier latency
    }
  }
  __syncthreads();
  __threadfence();
}

// A-operand packing for V_WMMA 16x16x32 bf16, this file's contiguous k order (B loads use the same):
//   element e in [0,512): lane l = e>>4, vgpr v = (e>>1)&7, half p = e&1
//   row m = l&15 ; k = ((l&16)?16:0) + v*2 + p
__device__ __forceinline__ void a_decode(int e, int& m, int& k) {
  int l = e >> 4;
  int v = (e >> 1) & 7;
  int p = e & 1;
  m = l & 15;
  k = ((l & 16) ? 16 : 0) + (v << 1) + p;
}

// ---------------- prep kernels (run every launch; graph-safe) ----------------
__global__ void prep_state(char* ws, const float* b_ih, const float* b_hh,
                           const float* b_nh1, const float* b_nh2, const float* b_nin1) {
  float* V1g   = (float*)(ws + OFF_V1G);
  float* cbuf  = (float*)(ws + OFF_CBUF);
  float* bsum  = (float*)(ws + OFF_BSUM);
  float* biasn = (float*)(ws + OFF_BIASN);
  float* zraw  = (float*)(ws + OFF_ZRAW);
  int i = blockIdx.x * blockDim.x + threadIdx.x;
  if (i >= 32768) return;
  zraw[i] = 0.0f;
  if (i < 1088) V1g[i] = (i < 512) ? 1.0f : ((i >= 514 && i < 1026) ? 1.0f : 0.0f);
  if (i < 1024) cbuf[i] = (i < 512) ? 1.0f : 0.0f;
  if (i < 2048) bsum[i] = b_ih[i] + b_hh[i];
  if (i < 1536) biasn[i] = (i < 512) ? b_nh1[i] : ((i < 1024) ? b_nh2[i - 512] : b_nin1[i - 1024]);
  if (i == 0) {
    ((unsigned*)ws)[0] = 0u;           // grid barrier counter
    ((unsigned*)ws)[1] = 0x007fffffu;  // enc(-inf) for atomic max
    ((float*)ws)[2]    = 0.0f;         // softmax denom
  }
}

__global__ void pack_p1(char* ws, const float* W_ih, const float* W_hh) {
  size_t idx = (size_t)blockIdx.x * blockDim.x + threadIdx.x;
  if (idx >= P1_ELEMS) return;
  __bf16* dst = (__bf16*)(ws + OFF_P1);
  int e = (int)(idx & 511);
  int tile = (int)(idx >> 9);
  int kt = tile % NKT1, rt = tile / NKT1;
  int m, k;
  a_decode(e, m, k);
  int row = rt * 16 + m;
  int kg = kt * 32 + k;          // combined K: [x(514) | h(512) | pad]
  float val = 0.0f;
  if (kg < 514)       val = W_ih[(size_t)row * 514 + kg];
  else if (kg < 1026) val = W_hh[(size_t)row * 512 + (kg - 514)];
  dst[idx] = (__bf16)val;
}

__global__ void pack_p2(char* ws, const float* W_nh1, const float* W_nh2, const float* W_nin1) {
  size_t idx = (size_t)blockIdx.x * blockDim.x + threadIdx.x;
  if (idx >= P2_ELEMS) return;
  __bf16* dst = (__bf16*)(ws + OFF_P2);
  int e = (int)(idx & 511);
  int tile = (int)(idx >> 9);
  int kt = tile % NKT2, rt = tile / NKT2;
  int m, k;
  a_decode(e, m, k);
  int row = rt * 16 + m;         // stacked rows: [h_n | c_n | zi_h]
  int kg = kt * 32 + k;
  float val;
  if (row < 512)       val = W_nh1[(size_t)row * 512 + kg];
  else if (row < 1024) val = W_nh2[(size_t)(row - 512) * 512 + kg];
  else                 val = W_nin1[(size_t)(row - 1024) * 512 + kg];
  dst[idx] = (__bf16)val;
}

// ---------------- persistent LSTM kernel ----------------
__global__ __launch_bounds__(NTHR, 1) void lstm_persist(char* ws, const float* W_out,
                                                        const int* nsp, float* out) {
  // LDS-resident weights: phase1 4 tiles x 34 kt, phase2 3 tiles x 16 kt (~190 KB total)
  __shared__ __align__(16) __bf16 aP1s[4][NKT1][512];
  __shared__ __align__(16) __bf16 aP2s[3][NKT2][512];
  __shared__ __align__(64) __bf16 v1b[KP1];
  __shared__ __align__(64) __bf16 hob[HDIM];
  __shared__ __align__(64) __bf16 cob[HDIM];
  __shared__ float part1[8][16];
  __shared__ float part2[8][16];
  __shared__ float zish[64];
  __shared__ float red[NTHR];

  const int tid  = threadIdx.x;
  const int wg   = blockIdx.x;
  const int wave = tid >> 5;
  const int lane = tid & 31;

  unsigned* bar   = (unsigned*)ws;
  unsigned* zmaxu = (unsigned*)ws + 1;
  float*    zsum  = (float*)ws + 2;
  float* V1g   = (float*)(ws + OFF_V1G);
  float* cbuf  = (float*)(ws + OFF_CBUF);
  float* gates = (float*)(ws + OFF_GATES);
  float* bsum  = (float*)(ws + OFF_BSUM);
  float* biasn = (float*)(ws + OFF_BIASN);
  float* zraw  = (float*)(ws + OFF_ZRAW);

  const int   T    = *nsp;
  const float invT = 1.0f / (float)T;

  // one-time copy of this WG's weight slice into LDS (L2 -> LDS, 16B vectors)
  {
    const uint4* s1 = (const uint4*)((const char*)ws + OFF_P1 + (size_t)wg * 4 * NKT1 * 512 * 2);
    uint4* d1 = (uint4*)&aP1s[0][0][0];
    for (int i = tid; i < 4 * NKT1 * 512 / 8; i += NTHR) d1[i] = s1[i];
    const uint4* s2 = (const uint4*)((const char*)ws + OFF_P2 + (size_t)wg * 3 * NKT2 * 512 * 2);
    uint4* d2 = (uint4*)&aP2s[0][0][0];
    for (int i = tid; i < 3 * NKT2 * 512 / 8; i += NTHR) d2[i] = s2[i];
  }
  __syncthreads();

  unsigned bn = 0;

  for (int step = 0; step < T; ++step) {
    // ---- stage state vector [x;h] into LDS as bf16 (B operand source) ----
    for (int k = tid; k < KP1; k += NTHR) v1b[k] = (__bf16)V1g[k];
    __syncthreads();

    // ---- phase 1: gates = Wg @ [x;h]  (WMMA matvec, vector broadcast over B cols) ----
    {
      const int tileL = wave >> 1;
      const int kh = wave & 1;
      // dual accumulators: break the WMMA->WMMA RAW chain on D (only 2 waves/SIMD here)
      v8f accA = {};
      v8f accB = {};
      const int kt0 = kh * 17;
      for (int kt = kt0; kt < kt0 + 16; kt += 2) {
        v16bf A0 = *(const v16bf*)&aP1s[tileL][kt][lane * 16];
        v16bf B0 = *(const v16bf*)&v1b[kt * 32 + (lane & 16)];
        v16bf A1 = *(const v16bf*)&aP1s[tileL][kt + 1][lane * 16];
        v16bf B1 = *(const v16bf*)&v1b[(kt + 1) * 32 + (lane & 16)];
        accA = __builtin_amdgcn_wmma_f32_16x16x32_bf16(false, A0, false, B0, (short)0, accA,
                                                       false, false);
        accB = __builtin_amdgcn_wmma_f32_16x16x32_bf16(false, A1, false, B1, (short)0, accB,
                                                       false, false);
      }
      { // 17th (odd) kt
        const int kt = kt0 + 16;
        v16bf A = *(const v16bf*)&aP1s[tileL][kt][lane * 16];
        v16bf B = *(const v16bf*)&v1b[kt * 32 + (lane & 16)];
        accA = __builtin_amdgcn_wmma_f32_16x16x32_bf16(false, A, false, B, (short)0, accA,
                                                       false, false);
      }
      if ((lane & 15) == 0) {               // all B cols equal -> lane 0 rows 0-7, lane 16 rows 8-15
        int half = lane >> 4;
        #pragma unroll
        for (int r = 0; r < 8; ++r) part1[wave][half * 8 + r] = accA[r] + accB[r];
      }
    }
    __syncthreads();
    if (tid < 64) {  // combine K halves, add bias, publish gates
      int tl = tid >> 4, r = tid & 15;
      int rowG = (wg * 4 + tl) * 16 + r;
      gates[rowG] = part1[tl * 2][r] + part1[tl * 2 + 1][r] + bsum[rowG];
    }
    ++bn; gbar(bar, bn * NWG);   // barrier A: gates visible

    // ---- pointwise LSTM (each WG computes full 512 redundantly into its LDS) ----
    const float* crd = cbuf + (step & 1) * HDIM;
    float*       cwr = cbuf + ((step + 1) & 1) * HDIM;
    for (int j = tid; j < HDIM; j += NTHR) {
      float ig = gates[j], fg = gates[512 + j], gg = gates[1024 + j], og = gates[1536 + j];
      float cn = sigm(fg) * crd[j] + sigm(ig) * tanhf(gg);
      float hn = sigm(og) * tanhf(cn);
      cob[j] = (__bf16)cn;
      hob[j] = (__bf16)hn;
      if (wg == 0) V1g[j] = hn;                   // x_n = [h_out, xy]
    }
    if (wg == 0 && tid == 0) V1g[513] = (float)(step + 1) * invT;  // V1g[512] stays 0
    __syncthreads();

    // ---- phase 2: [h_n; c_n; zi_h] = relu(Wn @ {h_out|c_out} + b) ----
    if (wave < 6) {
      const int tileL = wave >> 1;
      const int kh = wave & 1;
      const int rtG = wg * 3 + tileL;
      const __bf16* bsrc = (rtG < 32 || rtG >= 64) ? hob : cob;
      v8f accA = {};
      v8f accB = {};
      const int kt0 = kh * 8;
      for (int kt = kt0; kt < kt0 + 8; kt += 2) {
        v16bf A0 = *(const v16bf*)&aP2s[tileL][kt][lane * 16];
        v16bf B0 = *(const v16bf*)&bsrc[kt * 32 + (lane & 16)];
        v16bf A1 = *(const v16bf*)&aP2s[tileL][kt + 1][lane * 16];
        v16bf B1 = *(const v16bf*)&bsrc[(kt + 1) * 32 + (lane & 16)];
        accA = __builtin_amdgcn_wmma_f32_16x16x32_bf16(false, A0, false, B0, (short)0, accA,
                                                       false, false);
        accB = __builtin_amdgcn_wmma_f32_16x16x32_bf16(false, A1, false, B1, (short)0, accB,
                                                       false, false);
      }
      if ((lane & 15) == 0) {
        int half = lane >> 4;
        #pragma unroll
        for (int r = 0; r < 8; ++r) part2[wave][half * 8 + r] = accA[r] + accB[r];
      }
    }
    __syncthreads();
    if (tid < 48) {
      int tl = tid >> 4, r = tid & 15;
      int rowG = (wg * 3 + tl) * 16 + r;
      float val = part2[tl * 2][r] + part2[tl * 2 + 1][r] + biasn[rowG];
      val = fmaxf(val, 0.0f);
      float zc = 0.0f;
      if (rowG < 512)        V1g[514 + rowG] = val;        // h_n -> next step's h
      else if (rowG < 1024)  cwr[rowG - 512] = val;        // c_n (ping-pong)
      else {
        int j = rowG - 1024;                               // zi_h
        out[(size_t)T + (size_t)step * HDIM + j] = val;    // reference "outputs"
        zc = W_out[j] * val;                               // partial zi (b_out cancels in softmax)
      }
      zish[tid] = zc;
    }
    __syncthreads();
    if (wg >= 21) {                    // only WGs owning zi rows contribute
      if (tid < 16) zish[tid] = zish[tid] + zish[tid + 16] + zish[tid + 32];
      __syncthreads();
      if (tid < 4)  zish[tid] = zish[tid] + zish[tid + 4] + zish[tid + 8] + zish[tid + 12];
      __syncthreads();
      if (tid == 0) {
        float s = zish[0] + zish[1] + zish[2] + zish[3];
        __hip_atomic_fetch_add(&zraw[step], s, __ATOMIC_RELAXED, __HIP_MEMORY_SCOPE_AGENT);
      }
      __syncthreads();
    }
    ++bn; gbar(bar, bn * NWG);   // barrier B: state for next step visible
  }

  // ---- softmax(zraw) -> out[0..T) ----
  {
    float m = -3.4e38f;
    for (int t = wg * NTHR + tid; t < T; t += NWG * NTHR) m = fmaxf(m, zraw[t]);
    red[tid] = m; __syncthreads();
    for (int s = NTHR / 2; s > 0; s >>= 1) {
      if (tid < s) red[tid] = fmaxf(red[tid], red[tid + s]);
      __syncthreads();
    }
    if (tid == 0)
      __hip_atomic_fetch_max(zmaxu, fenc(red[0]), __ATOMIC_RELAXED, __HIP_MEMORY_SCOPE_AGENT);
    ++bn; gbar(bar, bn * NWG);

    float zm = fdec(__hip_atomic_load(zmaxu, __ATOMIC_RELAXED, __HIP_MEMORY_SCOPE_AGENT));
    float s = 0.0f;
    for (int t = wg * NTHR + tid; t < T; t += NWG * NTHR) s += expf(zraw[t] - zm);
    red[tid] = s; __syncthreads();
    for (int q = NTHR / 2; q > 0; q >>= 1) {
      if (tid < q) red[tid] += red[tid + q];
      __syncthreads();
    }
    if (tid == 0)
      __hip_atomic_fetch_add(zsum, red[0], __ATOMIC_RELAXED, __HIP_MEMORY_SCOPE_AGENT);
    ++bn; gbar(bar, bn * NWG);

    float inv = 1.0f / __hip_atomic_load(zsum, __ATOMIC_RELAXED, __HIP_MEMORY_SCOPE_AGENT);
    for (int t = wg * NTHR + tid; t < T; t += NWG * NTHR)
      out[t] = expf(zraw[t] - zm) * inv;
  }
}

// ---------------- host entry ----------------
extern "C" void kernel_launch(void* const* d_in, const int* in_sizes, int n_in,
                              void* d_out, int out_size, void* d_ws, size_t ws_size,
                              hipStream_t stream) {
  (void)in_sizes; (void)n_in; (void)out_size; (void)ws_size;
  const float* W_ih  = (const float*)d_in[0];
  const float* W_hh  = (const float*)d_in[1];
  const float* b_ih  = (const float*)d_in[2];
  const float* b_hh  = (const float*)d_in[3];
  const float* W_nh1 = (const float*)d_in[4];
  const float* b_nh1 = (const float*)d_in[5];
  const float* W_nh2 = (const float*)d_in[6];
  const float* b_nh2 = (const float*)d_in[7];
  const float* W_nin1= (const float*)d_in[8];
  const float* b_nin1= (const float*)d_in[9];
  const float* W_out = (const float*)d_in[10];
  // d_in[11] = b_out: constant shift inside softmax -> cancels, unused.
  const int*   nsp   = (const int*)d_in[12];
  // d_in[13] = root_leaves: level_frac = 0/(rl-2) = 0, unused.

  char*  ws  = (char*)d_ws;
  float* out = (float*)d_out;

  prep_state<<<32768 / NTHR, NTHR, 0, stream>>>(ws, b_ih, b_hh, b_nh1, b_nh2, b_nin1);
  pack_p1<<<(unsigned)((P1_ELEMS + NTHR - 1) / NTHR), NTHR, 0, stream>>>(ws, W_ih, W_hh);
  pack_p2<<<(unsigned)((P2_ELEMS + NTHR - 1) / NTHR), NTHR, 0, stream>>>(ws, W_nh1, W_nh2, W_nin1);
  lstm_persist<<<NWG, NTHR, 0, stream>>>(ws, W_out, nsp, out);
}
